// Qwen3MoeExperts_90666759618754
// MI455X (gfx1250) — matrix operand order repaired and checked by reading, not compiled, not run on hardware
//
#include <hip/hip_runtime.h>
#include <hip/hip_bf16.h>

#define E_ 32
#define H_ 1024
#define I_ 768
#define T_ 2048
#define K_ 8

typedef __attribute__((ext_vector_type(16))) __bf16 v16bf;
typedef __attribute__((ext_vector_type(8)))  __bf16 v8bf;
typedef __attribute__((ext_vector_type(4)))  __bf16 v4bf;
typedef __attribute__((ext_vector_type(8)))  float  v8f;
typedef __attribute__((ext_vector_type(4)))  float  v4f;

// Element position inside a 32-long per-row chunk for the 16-bit A-matrix
// 16x32 WMMA fragment (CDNA5 ISA 7.12.2):
//   half = (k>>3)&1, i = ((k&16)>>1)|(k&7), pos = half*16 + i
// For a 4-aligned k-run the 4 positions are contiguous.
__device__ __forceinline__ int a_pos(int k) {
    return (((k >> 3) & 1) << 4) | ((k & 16) >> 1) | (k & 7);
}

__global__ void k_zero(float* __restrict__ out, int n, int* __restrict__ cnt) {
    int i = blockIdx.x * blockDim.x + threadIdx.x;
    for (; i < n; i += gridDim.x * blockDim.x) out[i] = 0.f;
    if (blockIdx.x == 0 && threadIdx.x < E_) cnt[threadIdx.x] = 0;
}

__global__ void k_count(const int* __restrict__ sel, int* __restrict__ cnt) {
    int p = blockIdx.x * blockDim.x + threadIdx.x;
    if (p < T_ * K_) atomicAdd(&cnt[sel[p]], 1);
}

__global__ void k_scan(const int* __restrict__ cnt, int* __restrict__ offs,
                       int* __restrict__ cursor) {
    if (threadIdx.x == 0) {
        int acc = 0;
        for (int e = 0; e < E_; ++e) { offs[e] = acc; cursor[e] = acc; acc += cnt[e]; }
        offs[E_] = acc;
    }
}

__global__ void k_fill(const int* __restrict__ sel, const float* __restrict__ rw,
                       int* __restrict__ cursor, int* __restrict__ ids,
                       float* __restrict__ wts) {
    int p = blockIdx.x * blockDim.x + threadIdx.x;
    if (p < T_ * K_) {
        int e = sel[p];
        int pos = atomicAdd(&cursor[e], 1);
        ids[pos] = p / K_;
        wts[pos] = rw[p];
    }
}

__device__ __forceinline__ v4bf cvt4(v4f v) {
    v4bf r;
    r[0] = (__bf16)v[0]; r[1] = (__bf16)v[1];
    r[2] = (__bf16)v[2]; r[3] = (__bf16)v[3];
    return r;
}

__device__ __forceinline__ v8bf cvt8(v4f a, v4f b) {
    v8bf r;
#pragma unroll
    for (int j = 0; j < 4; ++j) { r[j] = (__bf16)a[j]; r[j + 4] = (__bf16)b[j]; }
    return r;
}

// Fused per-expert SwiGLU FFN, 32-token tile, 8 wave32 (256 threads).
// waves: wm = wave>>2 (token 16-row), wn = wave&3 (16-col of a 64-wide slab)
__global__ __launch_bounds__(256) void k_expert(
    const float* __restrict__ X,    // [T,H]
    const float* __restrict__ Wg,   // [E,I,H]
    const float* __restrict__ Wu,   // [E,I,H]
    const float* __restrict__ Wd,   // [E,H,I]
    const int*   __restrict__ offs,
    const int*   __restrict__ ids,
    const float* __restrict__ wts,
    float*       __restrict__ out)  // [T,H]
{
    __shared__ __align__(32) __bf16 sH[32 * I_];     // 48 KB: h tile, 32-k chunks, natural k order
    __shared__ __align__(32) __bf16 sX[32 * 32];     //  2 KB: X tile, row-major, natural k order
    // 8 KB weight staging: stage A = [gate(2048) | up(2048)],
    // stage B = one 64-wide Wd chunk as two 32-k sub-tiles
    __shared__ __align__(32) __bf16 sW[2 * 64 * 32];

    const int e    = blockIdx.y;
    const int base = offs[e] + blockIdx.x * 32;
    const int lim  = offs[e + 1];
    if (base >= lim) return;                       // block-uniform: EXEC stays full
    const int mcnt = (lim - base < 32) ? (lim - base) : 32;

    const int tid   = threadIdx.x;
    const int lane  = tid & 31;
    const int wave  = tid >> 5;
    const int wm    = wave >> 2;       // 0..1
    const int wn    = wave & 3;        // 0..3
    const int lhalf = lane >> 4;
    const int l15   = lane & 15;

    const float* wg_e = Wg + (size_t)e * I_ * H_;
    const float* wu_e = Wu + (size_t)e * I_ * H_;
    const float* wd_e = Wd + (size_t)e * H_ * I_;

    // ---- loop-invariant staging geometry ----
    const int sm   = tid >> 3;                       // X: token row 0..31
    const int sk4  = (tid & 7) << 2;                 // X: k offset 0,4,..,28
    const int mcl  = (sm < mcnt) ? sm : 0;           // clamp: garbage rows masked later
    const float* xrow = X + (size_t)ids[base + mcl] * H_ + sk4;
    const int xpos = (sm << 5) + sk4;                // contiguous 4-element slot
    const int sn   = tid >> 2;                       // W: row 0..63
    const int skw  = (tid & 3) << 3;                 // stage A: k offset 0,8,16,24
    const int wpos = (sn << 5) + skw;
    const int skw2 = (tid & 3) << 4;                 // stage B: k offset 0,16,32,48
    const int wpos2 = ((skw2 >> 5) << 11) + (sn << 5) + (skw2 & 31);
    const int aoff = ((wm * 16 + l15) << 5) + (lhalf << 4);
    const int boff = ((wn * 16 + l15) << 5) + (lhalf << 4);
    const int icol  = wn * 16 + l15;                 // h scatter column (fixed/lane)
    const int hcpos = icol & 31;
    const int hcbase= icol >> 5;
    const int m0    = wm * 16 + (lhalf << 3);        // C-fragment row base

    // ---- Stage A: h = silu(X Wg^T) * (X Wu^T), tile [32 x I] kept in LDS ----
    for (int i0 = 0; i0 < I_; i0 += 64) {
        const float* wgrow = wg_e + (size_t)(i0 + sn) * H_ + skw;
        const float* wurow = wu_e + (size_t)(i0 + sn) * H_ + skw;
        v8f ag = {}; v8f au = {};
        // software pipeline: registers hold k-step data loaded one iteration ahead
        v4f rx  = *(const v4f*)(xrow);
        v4f rg0 = *(const v4f*)(wgrow),     rg1 = *(const v4f*)(wgrow + 4);
        v4f ru0 = *(const v4f*)(wurow),     ru1 = *(const v4f*)(wurow + 4);
        for (int k0 = 0; k0 < H_; k0 += 32) {
            __syncthreads();                              // prev frag reads done
            *(v4bf*)&sX[xpos]        = cvt4(rx);
            *(v8bf*)&sW[wpos]        = cvt8(rg0, rg1);    // gate half
            *(v8bf*)&sW[2048 + wpos] = cvt8(ru0, ru1);    // up half
            if (k0 + 32 < H_) {                           // issue next step's loads
                rx  = *(const v4f*)(xrow  + k0 + 32);
                rg0 = *(const v4f*)(wgrow + k0 + 32); rg1 = *(const v4f*)(wgrow + k0 + 36);
                ru0 = *(const v4f*)(wurow + k0 + 32); ru1 = *(const v4f*)(wurow + k0 + 36);
            }
            if (k0 + 64 < H_) {                           // L2 prefetch 2 steps ahead
                __builtin_prefetch(wgrow + k0 + 64, 0, 3);
                __builtin_prefetch(wurow + k0 + 64, 0, 3);
            }
            __syncthreads();                              // staging visible
            v16bf a  = *(const v16bf*)&sX[aoff];
            v16bf bg = *(const v16bf*)&sW[boff];
            v16bf bu = *(const v16bf*)&sW[2048 + boff];
            ag = __builtin_amdgcn_wmma_f32_16x16x32_bf16(false, a, false, bg,
                                                         (short)0, ag, false, false);
            au = __builtin_amdgcn_wmma_f32_16x16x32_bf16(false, a, false, bu,
                                                         (short)0, au, false, false);
        }
        // SwiGLU + scatter into sH (32-k chunks along I, natural k order)
        const int c = (i0 >> 5) + hcbase;
#pragma unroll
        for (int r = 0; r < 8; ++r) {
            float g  = ag[r];
            float hv = (g / (1.f + __expf(-g))) * au[r];
            sH[((c * 32 + m0 + r) << 5) + hcpos] = (__bf16)hv;
        }
    }
    __syncthreads();

    // ---- preload per-lane output token ids / routing weights (loop-invariant) ----
    int   ot[8];
    float ow[8];
#pragma unroll
    for (int r = 0; r < 8; ++r) {
        int m = m0 + r;
        bool v = (m < mcnt);
        ot[r] = v ? ids[base + m] : -1;
        ow[r] = v ? wts[base + m] : 0.f;
    }

    // ---- Stage B: out[t,:] += w[t] * (h Wd^T), 64-wide k-chunks: 2 WMMA/sync ----
    for (int h0 = 0; h0 < H_; h0 += 64) {
        const float* wdrow = wd_e + (size_t)(h0 + sn) * I_ + skw2;
        v8f ad = {};
        v4f rd0 = *(const v4f*)(wdrow),     rd1 = *(const v4f*)(wdrow + 4);
        v4f rd2 = *(const v4f*)(wdrow + 8), rd3 = *(const v4f*)(wdrow + 12);
        for (int k0 = 0; k0 < I_; k0 += 64) {
            __syncthreads();
            *(v8bf*)&sW[wpos2]     = cvt8(rd0, rd1);
            *(v8bf*)&sW[wpos2 + 8] = cvt8(rd2, rd3);
            if (k0 + 64 < I_) {
                rd0 = *(const v4f*)(wdrow + k0 + 64); rd1 = *(const v4f*)(wdrow + k0 + 68);
                rd2 = *(const v4f*)(wdrow + k0 + 72); rd3 = *(const v4f*)(wdrow + k0 + 76);
            }
            if (k0 + 128 < I_) __builtin_prefetch(wdrow + k0 + 128, 0, 3);
            __syncthreads();
            int c = k0 >> 5;
            v16bf a0 = *(const v16bf*)&sH[(c << 10) + aoff];
            v16bf a1 = *(const v16bf*)&sH[((c + 1) << 10) + aoff];
            v16bf b0 = *(const v16bf*)&sW[boff];
            v16bf b1 = *(const v16bf*)&sW[2048 + boff];
            ad = __builtin_amdgcn_wmma_f32_16x16x32_bf16(false, a0, false, b0,
                                                         (short)0, ad, false, false);
            ad = __builtin_amdgcn_wmma_f32_16x16x32_bf16(false, a1, false, b1,
                                                         (short)0, ad, false, false);
        }
#pragma unroll
        for (int r = 0; r < 8; ++r) {
            if (ot[r] >= 0) {
                int n = h0 + wn * 16 + l15;
                atomicAdd(&out[(size_t)ot[r] * H_ + n], ow[r] * ad[r]);
            }
        }
    }
}

extern "C" void kernel_launch(void* const* d_in, const int* in_sizes, int n_in,
                              void* d_out, int out_size, void* d_ws, size_t ws_size,
                              hipStream_t stream) {
    const float* X   = (const float*)d_in[0];
    const float* rw  = (const float*)d_in[1];
    const int*   sel = (const int*)d_in[2];
    const float* Wg  = (const float*)d_in[3];
    const float* Wu  = (const float*)d_in[4];
    const float* Wd  = (const float*)d_in[5];
    float* out = (float*)d_out;

    char* ws    = (char*)d_ws;
    int*   cnt    = (int*)(ws + 0);
    int*   cursor = (int*)(ws + 128);
    int*   offs   = (int*)(ws + 256);
    int*   ids    = (int*)(ws + 512);
    float* wts    = (float*)(ws + 512 + (size_t)T_ * K_ * 4);

    k_zero <<<512, 256, 0, stream>>>(out, T_ * H_, cnt);
    k_count<<<(T_ * K_ + 255) / 256, 256, 0, stream>>>(sel, cnt);
    k_scan <<<1, 32, 0, stream>>>(cnt, offs, cursor);
    k_fill <<<(T_ * K_ + 255) / 256, 256, 0, stream>>>(sel, rw, cursor, ids, wts);
    // worst case one expert owns all T*K pairs -> 512 tiles of 32; empty tiles exit early
    k_expert<<<dim3(512, E_), 256, 0, stream>>>(X, Wg, Wu, Wd, offs, ids, wts, out);
}
